// MultiScaleMemoryAttention_15109694947380
// MI455X (gfx1250) — compile-verified, matrix operand order checked by reading, not run on hardware
//
#include <hip/hip_runtime.h>

// ---------------- problem constants ----------------
#define BB 2
#define TT 2048
#define DDIM 768
#define HH 12
#define BT (BB*TT)            // 4096
#define NPROJ 3328            // padded fused-projection width (26*128)
// fused projection column (== packed weight row) offsets
#define C_Q    0
#define C_K    768
#define C_V    1536
#define C_W1W  2304
#define C_W2W  2352
#define C_W1R  2400
#define C_W2R  2448
#define C_MEMV 2496
#define C_MEMG 3264
#define C_USED 3276
#define DECAYF 0.99f
#define NEGBIG (-1.0e30f)
#define ASTRIDE 40            // LDS tile row stride in halves (80B: 16B-aligned, 20-bank skew)

typedef __attribute__((ext_vector_type(8)))  _Float16 v8h;
typedef __attribute__((ext_vector_type(16))) _Float16 v16h;
typedef __attribute__((ext_vector_type(8)))  float    v8f;

// ---- A/B fragment loader for V_WMMA_F32_16X16X32_F16 (wave32) ----
// lane<16 : elements 0..7 = K[kb+0..7],  8..15 = K[kb+16..23]  (row/col = lane)
// lane>=16: elements 0..7 = K[kb+8..15], 8..15 = K[kb+24..31]  (row/col = lane-16)
static __device__ __forceinline__ v16h load_frag(const _Float16* p, int g) {
  v8h lo = *(const v8h*)(p + 8*g);
  v8h hi = *(const v8h*)(p + 16 + 8*g);
  return __builtin_shufflevector(lo, hi, 0,1,2,3,4,5,6,7,8,9,10,11,12,13,14,15);
}

static __device__ __forceinline__ v8f wmma16(v16h a, v16h b, v8f c) {
  return __builtin_amdgcn_wmma_f32_16x16x32_f16(false, a, false, b, (short)0, c, false, false);
}

// 16-byte async global->LDS copy (per-lane addresses), tracked by ASYNCcnt.
static __device__ __forceinline__ void async_copy_b128(const _Float16* gsrc, _Float16* ldst) {
  uint32_t l = (uint32_t)(uintptr_t)ldst;      // low 32 bits of flat shared ptr == LDS offset
  uint64_t a = (uint64_t)(uintptr_t)gsrc;
  asm volatile("global_load_async_to_lds_b128 %0, %1, off"
               :: "v"(l), "v"(a) : "memory");
}

static __device__ __forceinline__ void wait_async_all() {
  asm volatile("s_wait_asynccnt 0x0" ::: "memory");
}

static __device__ __forceinline__ float sigmoidf_(float x) {
  return 1.0f / (1.0f + __expf(-x));
}

static __device__ __forceinline__ void plucker6(const float* p1, const float* p2, float* L) {
  L[0] = p1[0]*p2[1] - p1[1]*p2[0];
  L[1] = p1[0]*p2[2] - p1[2]*p2[0];
  L[2] = p1[0]*p2[3] - p1[3]*p2[0];
  L[3] = p1[1]*p2[2] - p1[2]*p2[1];
  L[4] = p1[1]*p2[3] - p1[3]*p2[1];
  L[5] = p1[2]*p2[3] - p1[3]*p2[2];
}

// ---------------- tiny utility kernels ----------------
__global__ void k_cvt_h(_Float16* __restrict__ dst, const float* __restrict__ src, int n) {
  int i = blockIdx.x * 256 + threadIdx.x;
  if (i < n) dst[i] = (_Float16)src[i];
}

__global__ void k_zero_h(_Float16* __restrict__ dst, int n) {
  int i = blockIdx.x * 256 + threadIdx.x;
  if (i < n) dst[i] = (_Float16)0.0f;
}

__global__ void k_pack_bias(float* __restrict__ dst,
                            const float* __restrict__ qkv_b,
                            const float* __restrict__ memv_b,
                            const float* __restrict__ memg_b) {
  int i = blockIdx.x * 256 + threadIdx.x;
  if (i >= NPROJ) return;
  float v = 0.0f;
  if (i < 2304)                       v = qkv_b[i];
  else if (i >= C_MEMV && i < C_MEMG) v = memv_b[i - C_MEMV];
  else if (i >= C_MEMG && i < C_USED) v = memg_b[i - C_MEMG];
  dst[i] = v;
}

// Build Vt[bh][d][t] (f16) from fused projection (transposed V for PV WMMA B-frags)
__global__ void k_build_vt(const _Float16* __restrict__ proj, _Float16* __restrict__ Vt) {
  int idx = blockIdx.x * 256 + threadIdx.x;             // over B*H*64*T
  if (idx >= BB*HH*64*TT) return;
  int t  = idx % TT;
  int rs = idx / TT;
  int d  = rs % 64;
  int bh = rs / 64;
  int b = bh / HH, h = bh % HH;
  Vt[idx] = proj[(size_t)(b*TT + t) * NPROJ + C_V + h*64 + d];
}

// ---------------- C = A * B^T + bias GEMM (f16 in, WMMA, f32 acc) ------------
// Async-LDS double-buffered: 128x128 WG tile, k-step 32, 8 waves as 4(M)x2(N),
// wave tile 32x64. Tiles staged with GLOBAL_LOAD_ASYNC_TO_LDS_B128 (ASYNCcnt),
// fragments fed to WMMA from LDS via ds_load_b128.
__global__ __launch_bounds__(256)
void gemm_wmma(const _Float16* __restrict__ A, int lda,
               const _Float16* __restrict__ Bw, int ldb,
               const float* __restrict__ bias,
               float* __restrict__ Cf, _Float16* __restrict__ Ch, int ldc, int K) {
  __shared__ __align__(16) _Float16 sA[2][128 * ASTRIDE];
  __shared__ __align__(16) _Float16 sB[2][128 * ASTRIDE];
  const int tid  = threadIdx.x;
  const int lane = tid & 31;
  const int g = lane >> 4, r = lane & 15;
  const int wave = tid >> 5;
  const int wm = wave >> 1, wn = wave & 1;
  const int m0 = blockIdx.y * 128;
  const int n0 = blockIdx.x * 128;

  // per-thread async copy assignment: rows (crow, crow+64), 16B chunk ckp
  const int crow = tid >> 2;
  const int ckp  = (tid & 3) * 8;

  auto issue_tile = [&](int buf, int kb) {
    const _Float16* gA = A  + (size_t)(m0 + crow) * lda + kb + ckp;
    const _Float16* gB = Bw + (size_t)(n0 + crow) * ldb + kb + ckp;
    _Float16* lA = &sA[buf][crow * ASTRIDE + ckp];
    _Float16* lB = &sB[buf][crow * ASTRIDE + ckp];
    async_copy_b128(gA,                     lA);
    async_copy_b128(gA + (size_t)64 * lda,  lA + 64 * ASTRIDE);
    async_copy_b128(gB,                     lB);
    async_copy_b128(gB + (size_t)64 * ldb,  lB + 64 * ASTRIDE);
  };

  const int nk = K >> 5;
  issue_tile(0, 0);
  wait_async_all();
  __syncthreads();

  v8f c[2][4] = {};
  for (int i = 0; i < nk; ++i) {
    if (i + 1 < nk) issue_tile((i + 1) & 1, (i + 1) << 5);   // overlap with WMMAs below
    const _Float16* tA = sA[i & 1];
    const _Float16* tB = sB[i & 1];
    v16h a[2], b[4];
#pragma unroll
    for (int mt = 0; mt < 2; ++mt)
      a[mt] = load_frag(tA + (wm*32 + mt*16 + r) * ASTRIDE, g);
#pragma unroll
    for (int nt = 0; nt < 4; ++nt)
      b[nt] = load_frag(tB + (wn*64 + nt*16 + r) * ASTRIDE, g);
#pragma unroll
    for (int mt = 0; mt < 2; ++mt)
#pragma unroll
      for (int nt = 0; nt < 4; ++nt)
        c[mt][nt] = wmma16(a[mt], b[nt], c[mt][nt]);
    wait_async_all();      // next tile landed (had the whole WMMA block to do so)
    __syncthreads();
  }

#pragma unroll
  for (int mt = 0; mt < 2; ++mt)
#pragma unroll
    for (int nt = 0; nt < 4; ++nt)
#pragma unroll
      for (int rr = 0; rr < 8; ++rr) {
        int m = m0 + wm*32 + mt*16 + rr + 8*g;
        int n = n0 + wn*64 + nt*16 + r;
        float v = c[mt][nt][rr] + bias[n];
        if (Cf) Cf[(size_t)m * ldc + n] = v;
        if (Ch) Ch[(size_t)m * ldc + n] = (_Float16)v;
      }
}

// ---------------- flash attention (causal), WMMA both GEMMs ----------------
// grid (B*H, T/128), block 128 (4 waves); wave handles 32 query rows
__global__ __launch_bounds__(128)
void attn_kernel(const _Float16* __restrict__ proj, const _Float16* __restrict__ Vt,
                 float* __restrict__ seq_out) {
  const int bh = blockIdx.x, b = bh / HH, h = bh % HH;
  const int lane = threadIdx.x & 31, wave = threadIdx.x >> 5;
  const int g = lane >> 4, r = lane & 15;
  const int t0 = blockIdx.y * 128 + wave * 32;
  const _Float16* base = proj + (size_t)b * TT * NPROJ;
  __shared__ __align__(16) _Float16 Plds[4][32 * 64];
  _Float16* P = Plds[wave];

  // Q fragments, pre-scaled by dh^-0.5 = 0.125 (exact in f16)
  v16h qa[2][2];
#pragma unroll
  for (int mt = 0; mt < 2; ++mt)
#pragma unroll
    for (int ks = 0; ks < 2; ++ks) {
      v16h q = load_frag(base + (size_t)(t0 + mt*16 + r) * NPROJ + C_Q + h*64 + ks*32, g);
#pragma unroll
      for (int i = 0; i < 16; ++i) q[i] = q[i] * (_Float16)0.125f;
      qa[mt][ks] = q;
    }

  float mrow[2][8], lrow[2][8];
#pragma unroll
  for (int mt = 0; mt < 2; ++mt)
#pragma unroll
    for (int rr = 0; rr < 8; ++rr) { mrow[mt][rr] = NEGBIG; lrow[mt][rr] = 0.0f; }
  v8f o[2][4] = {};

  const int ktmax = (t0 + 31) >> 6;
  for (int kt = 0; kt <= ktmax; ++kt) {
    const int s0 = kt * 64;
    if (kt < ktmax)  // near-scope prefetch of next K tile rows
      __builtin_prefetch(base + (size_t)(s0 + 64 + r) * NPROJ + C_K + h*64, 0, 3);
    const bool need_mask = (s0 + 63 > t0);
#pragma unroll
    for (int mt = 0; mt < 2; ++mt) {
      v8f s[4] = {};
#pragma unroll
      for (int ks = 0; ks < 2; ++ks)
#pragma unroll
        for (int nt = 0; nt < 4; ++nt) {
          v16h kb = load_frag(base + (size_t)(s0 + nt*16 + r) * NPROJ + C_K + h*64 + ks*32, g);
          s[nt] = wmma16(qa[mt][ks], kb, s[nt]);
        }
      if (need_mask) {
#pragma unroll
        for (int nt = 0; nt < 4; ++nt)
#pragma unroll
          for (int rr = 0; rr < 8; ++rr)
            if (s0 + nt*16 + r > t0 + mt*16 + rr + 8*g) s[nt][rr] = NEGBIG;
      }
      // online softmax update per row (rows are half-wave-uniform)
#pragma unroll
      for (int rr = 0; rr < 8; ++rr) {
        float v = fmaxf(fmaxf(s[0][rr], s[1][rr]), fmaxf(s[2][rr], s[3][rr]));
        v = fmaxf(v, __shfl_xor(v, 1, 32));
        v = fmaxf(v, __shfl_xor(v, 2, 32));
        v = fmaxf(v, __shfl_xor(v, 4, 32));
        v = fmaxf(v, __shfl_xor(v, 8, 32));
        float mnew  = fmaxf(mrow[mt][rr], v);
        float alpha = __expf(mrow[mt][rr] - mnew);
        mrow[mt][rr] = mnew;
        float rs = 0.0f;
#pragma unroll
        for (int nt = 0; nt < 4; ++nt) {
          float pv = __expf(s[nt][rr] - mnew);
          rs += pv;
          P[(mt*16 + rr + 8*g) * 64 + nt*16 + r] = (_Float16)pv;
        }
        rs += __shfl_xor(rs, 1, 32);
        rs += __shfl_xor(rs, 2, 32);
        rs += __shfl_xor(rs, 4, 32);
        rs += __shfl_xor(rs, 8, 32);
        lrow[mt][rr] = lrow[mt][rr] * alpha + rs;
#pragma unroll
        for (int nt = 0; nt < 4; ++nt) o[mt][nt][rr] *= alpha;
      }
    }
    // O += P * V   (P A-frags via LDS; V B-frags via pre-transposed Vt)
#pragma unroll
    for (int ks = 0; ks < 2; ++ks) {
      v16h pa[2];
#pragma unroll
      for (int mt = 0; mt < 2; ++mt)
        pa[mt] = load_frag(P + (mt*16 + r) * 64 + ks*32, g);
      v16h vb[4];
#pragma unroll
      for (int nt = 0; nt < 4; ++nt)
        vb[nt] = load_frag(Vt + ((size_t)bh * 64 + nt*16 + r) * TT + s0 + ks*32, g);
#pragma unroll
      for (int mt = 0; mt < 2; ++mt)
#pragma unroll
        for (int nt = 0; nt < 4; ++nt)
          o[mt][nt] = wmma16(pa[mt], vb[nt], o[mt][nt]);
    }
  }
#pragma unroll
  for (int mt = 0; mt < 2; ++mt)
#pragma unroll
    for (int rr = 0; rr < 8; ++rr) {
      float inv = 1.0f / lrow[mt][rr];
      int t = t0 + mt*16 + rr + 8*g;
#pragma unroll
      for (int nt = 0; nt < 4; ++nt)
        seq_out[(size_t)(b*TT + t) * DDIM + h*64 + nt*16 + r] = o[mt][nt][rr] * inv;
    }
}

// ---------------- Plucker precompute: read_h (unit 6-vec) and jw per offset ----
__global__ void k_exterior(const _Float16* __restrict__ proj,
                           _Float16* __restrict__ rh, _Float16* __restrict__ jwh) {
  int idx = blockIdx.x * 256 + threadIdx.x;   // over B*H*T
  if (idx >= BB*HH*TT) return;
  int t  = idx % TT;
  int bh = idx / TT;
  int h = bh % HH, b = bh / HH;
  const _Float16* row = proj + (size_t)(b*TT + t) * NPROJ;
  float p1[4], p2[4], L[6];
#pragma unroll
  for (int p = 0; p < 4; ++p) { p1[p] = (float)row[C_W1R + h*4 + p]; p2[p] = (float)row[C_W2R + h*4 + p]; }
  plucker6(p1, p2, L);
  float n2 = L[0]*L[0]+L[1]*L[1]+L[2]*L[2]+L[3]*L[3]+L[4]*L[4]+L[5]*L[5];
  float inv = 1.0f / fmaxf(sqrtf(n2), 1e-12f);
  _Float16* rdst = rh + (size_t)idx * 8;
#pragma unroll
  for (int j = 0; j < 6; ++j) rdst[j] = (_Float16)(L[j] * inv);
  rdst[6] = (_Float16)0.0f; rdst[7] = (_Float16)0.0f;

  float w2[4];
#pragma unroll
  for (int p = 0; p < 4; ++p) w2[p] = (float)row[C_W2W + h*4 + p];
  const int offs[4] = {1, 2, 4, 8};
#pragma unroll
  for (int k = 0; k < 4; ++k) {
    float w1[4] = {0.f, 0.f, 0.f, 0.f};
    if (t >= offs[k]) {
      const _Float16* rp = proj + (size_t)(b*TT + t - offs[k]) * NPROJ;
#pragma unroll
      for (int p = 0; p < 4; ++p) w1[p] = (float)rp[C_W1W + h*4 + p];
    }
    plucker6(w1, w2, L);
    n2 = L[0]*L[0]+L[1]*L[1]+L[2]*L[2]+L[3]*L[3]+L[4]*L[4]+L[5]*L[5];
    inv = 1.0f / fmaxf(sqrtf(n2), 1e-12f);
    float jw[6] = { L[5]*inv, -L[4]*inv, L[3]*inv, L[2]*inv, -L[1]*inv, L[0]*inv };
    _Float16* jd = jwh + ((size_t)(k*BB*HH + bh) * TT + t) * 8;
#pragma unroll
    for (int j = 0; j < 6; ++j) jd[j] = (_Float16)jw[j];
    jd[6] = (_Float16)0.0f; jd[7] = (_Float16)0.0f;
  }
}

// ---------------- decayed incidence row-sum via K=6 WMMA tiles ----------------
// grid (B*H, T/16), one wave per (bh, t-tile of 16 rows)
__global__ __launch_bounds__(32)
void memscore_kernel(const _Float16* __restrict__ rh, const _Float16* __restrict__ jwh,
                     float* __restrict__ mem_score) {
  const int bh = blockIdx.x;
  const int t0 = blockIdx.y * 16;
  const int lane = threadIdx.x & 31, g = lane >> 4, r = lane & 15;

  v16h a;
#pragma unroll
  for (int i = 0; i < 16; ++i) a[i] = (_Float16)0.0f;
  if (g == 0) {
    v8h lo = *(const v8h*)(rh + ((size_t)bh * TT + t0 + r) * 8);
#pragma unroll
    for (int i = 0; i < 8; ++i) a[i] = lo[i];
  }

  float acc[8] = {0,0,0,0,0,0,0,0};
  const float lanefac = __powf(DECAYF, (float)(8*g - r));   // decay^(m_hi - n) part
  float dr[8]; dr[0] = 1.0f;
#pragma unroll
  for (int rr = 1; rr < 8; ++rr) dr[rr] = dr[rr-1] * DECAYF;
  float base = __powf(DECAYF, (float)t0);                   // decay^(t0 - s0), s0=0
  const float inv16 = __powf(DECAYF, -16.0f);

  for (int s0 = 0; s0 <= t0; s0 += 16) {
    const bool diag = (s0 == t0);
#pragma unroll
    for (int off = 0; off < 4; ++off) {
      v16h bf;
#pragma unroll
      for (int i = 0; i < 16; ++i) bf[i] = (_Float16)0.0f;
      if (g == 0) {
        v8h lo = *(const v8h*)(jwh + ((size_t)(off*BB*HH + bh) * TT + s0 + r) * 8);
#pragma unroll
        for (int i = 0; i < 8; ++i) bf[i] = lo[i];
      }
      v8f c = {};
      c = wmma16(a, bf, c);   // c(m,n) = read_h[t0+m] . jw[s0+n]
#pragma unroll
      for (int rr = 0; rr < 8; ++rr) {
        float w = base * lanefac * dr[rr];                  // decay^(t-s)
        if (diag && (rr + 8*g) <= r) w = 0.0f;              // strict lower triangle
        float v = c[rr];
        acc[rr] += v * v * w;
      }
    }
    base *= inv16;
  }
#pragma unroll
  for (int rr = 0; rr < 8; ++rr) {
    float v = acc[rr];
    v += __shfl_xor(v, 1, 32);
    v += __shfl_xor(v, 2, 32);
    v += __shfl_xor(v, 4, 32);
    v += __shfl_xor(v, 8, 32);
    if (r == 0) mem_score[(size_t)bh * TT + t0 + rr + 8*g] = v * 0.25f;  // /n_active
  }
}

// ---------------- gate fuse: y = seq_out + gated * mem_val, emit f16 ----------
__global__ __launch_bounds__(256)
void fuse_kernel(const _Float16* __restrict__ proj, const float* __restrict__ seq_out,
                 const float* __restrict__ mem_score, const float* __restrict__ mem_scale,
                 _Float16* __restrict__ yh) {
  int bt = blockIdx.x;            // 0..BT-1
  int b = bt / TT, t = bt % TT;
  const _Float16* row = proj + (size_t)bt * NPROJ;
  float gsum = 0.0f;
#pragma unroll
  for (int hh = 0; hh < HH; ++hh) {
    float gate = sigmoidf_((float)row[C_MEMG + hh]);        // bias already folded in
    float ms   = mem_score[(size_t)(b*HH + hh) * TT + t];
    gsum += sigmoidf_(ms * mem_scale[hh]) * gate;
  }
  float gg = gsum * (1.0f / (float)HH);
  for (int d = threadIdx.x; d < DDIM; d += blockDim.x) {
    float y = seq_out[(size_t)bt * DDIM + d] + gg * (float)row[C_MEMV + d];
    yh[(size_t)bt * DDIM + d] = (_Float16)y;
  }
}

// ---------------- host-side orchestration ----------------
extern "C" void kernel_launch(void* const* d_in, const int* in_sizes, int n_in,
                              void* d_out, int out_size, void* d_ws, size_t ws_size,
                              hipStream_t stream) {
  (void)in_sizes; (void)n_in; (void)out_size; (void)ws_size;
  const float* x        = (const float*)d_in[0];
  const float* qkv_w    = (const float*)d_in[1];
  const float* qkv_b    = (const float*)d_in[2];
  const float* w1_write = (const float*)d_in[3];
  const float* w2_write = (const float*)d_in[4];
  const float* w1_read  = (const float*)d_in[5];
  const float* w2_read  = (const float*)d_in[6];
  const float* memv_w   = (const float*)d_in[7];
  const float* memv_b   = (const float*)d_in[8];
  const float* memg_w   = (const float*)d_in[9];
  const float* memg_b   = (const float*)d_in[10];
  const float* mem_scale= (const float*)d_in[11];
  const float* out_w    = (const float*)d_in[12];
  const float* out_b    = (const float*)d_in[13];
  float* out = (float*)d_out;

  char* p = (char*)d_ws;
  auto carve = [&](size_t bytes) { char* q = p; p += (bytes + 255) & ~(size_t)255; return q; };
  _Float16* xh     = (_Float16*)carve((size_t)BT * DDIM * 2);
  _Float16* Wh     = (_Float16*)carve((size_t)NPROJ * DDIM * 2);
  _Float16* outWh  = (_Float16*)carve((size_t)DDIM * DDIM * 2);
  float*    biasp  = (float*)   carve((size_t)NPROJ * 4);
  _Float16* proj   = (_Float16*)carve((size_t)BT * NPROJ * 2);
  _Float16* Vt     = (_Float16*)carve((size_t)BB*HH*64*TT * 2);
  _Float16* rh     = (_Float16*)carve((size_t)BB*HH*TT * 8 * 2);
  _Float16* jwh    = (_Float16*)carve((size_t)4*BB*HH*TT * 8 * 2);
  float*    mscore = (float*)   carve((size_t)BB*HH*TT * 4);
  float*    seqout = (float*)   carve((size_t)BT * DDIM * 4);
  _Float16* yh     = (_Float16*)carve((size_t)BT * DDIM * 2);

  auto cvt = [&](_Float16* dst, const float* src, int n) {
    k_cvt_h<<<(n + 255) / 256, 256, 0, stream>>>(dst, src, n);
  };
  // f16 conversions + packed-weight assembly
  cvt(xh, x, BT * DDIM);
  cvt(Wh + (size_t)0      * DDIM, qkv_w,    2304 * DDIM);
  cvt(Wh + (size_t)C_W1W  * DDIM, w1_write,   48 * DDIM);
  cvt(Wh + (size_t)C_W2W  * DDIM, w2_write,   48 * DDIM);
  cvt(Wh + (size_t)C_W1R  * DDIM, w1_read,    48 * DDIM);
  cvt(Wh + (size_t)C_W2R  * DDIM, w2_read,    48 * DDIM);
  cvt(Wh + (size_t)C_MEMV * DDIM, memv_w,    768 * DDIM);
  cvt(Wh + (size_t)C_MEMG * DDIM, memg_w,     12 * DDIM);
  {
    int npad = (NPROJ - C_USED) * DDIM;
    k_zero_h<<<(npad + 255) / 256, 256, 0, stream>>>(Wh + (size_t)C_USED * DDIM, npad);
  }
  cvt(outWh, out_w, DDIM * DDIM);
  k_pack_bias<<<(NPROJ + 255) / 256, 256, 0, stream>>>(biasp, qkv_b, memv_b, memg_b);

  // fused projection GEMM: proj[BT x NPROJ] = xh * Wh^T + biasp (f16 out)
  gemm_wmma<<<dim3(NPROJ / 128, BT / 128), 256, 0, stream>>>(
      xh, DDIM, Wh, DDIM, biasp, (float*)nullptr, proj, NPROJ, DDIM);

  // V transpose + Plucker precompute
  {
    int n = BB*HH*64*TT;
    k_build_vt<<<(n + 255) / 256, 256, 0, stream>>>(proj, Vt);
  }
  {
    int n = BB*HH*TT;
    k_exterior<<<(n + 255) / 256, 256, 0, stream>>>(proj, rh, jwh);
  }

  // causal flash attention
  attn_kernel<<<dim3(BB*HH, TT / 128), 128, 0, stream>>>(proj, Vt, seqout);

  // decayed Plucker incidence row-sums (WMMA K=6)
  memscore_kernel<<<dim3(BB*HH, TT / 16), 32, 0, stream>>>(rh, jwh, mscore);

  // gate + fuse -> f16 input of final GEMM
  fuse_kernel<<<BT, 256, 0, stream>>>(proj, seqout, mscore, mem_scale, yh);

  // final GEMM: out = yh * out_w^T + out_b (f32 out)
  gemm_wmma<<<dim3(DDIM / 128, BT / 128), 256, 0, stream>>>(
      yh, DDIM, outWh, DDIM, out_b, out, (_Float16*)nullptr, DDIM, DDIM);
}
